// LigandMPNN_42752104465253
// MI455X (gfx1250) — matrix operand order repaired and checked by reading, not compiled, not run on hardware
//
#include <hip/hip_runtime.h>
#include <hip/hip_bf16.h>
#include <cstdint>
#include <cstddef>

// ---------------------------------------------------------------------------
// LigandMPNN-style GNN for MI455X (gfx1250), wave32 + WMMA f16 16x16x32.
// GEMM weights are staged block-wide into LDS with CDNA5 async loads
// (global_load_async_to_lds_b128 + s_wait_asynccnt), then fed to
// v_wmma_f32_16x16x32_f16 via conflict-free ds_load_b128. K is a template
// constant so the k-loop fully unrolls (no branches, batched DS waits,
// back-to-back WMMA bursts). A-stream prefetched with global_prefetch_b8.
// Requires d_ws >= ~330 MB.
// ---------------------------------------------------------------------------

typedef __attribute__((ext_vector_type(16))) _Float16 v16h;
typedef __attribute__((ext_vector_type(8)))  _Float16 v8h;
typedef __attribute__((ext_vector_type(8)))  float    v8f;

#define BB    4
#define NNODE 1024
#define KNBR  32
#define NODES (BB * NNODE)      // 4096
#define EC    (NODES * KNBR)    // 131072

__device__ __forceinline__ float gelu_f(float x) {
  return 0.5f * x * (1.0f + erff(x * 0.70710678118654752f));
}

// ---------------------------------------------------------------------------
// 1. coords: N, CA, C, O from X; CB virtual atom -> coords (B,N,5,3)
// ---------------------------------------------------------------------------
__global__ void coords_kernel(const float* __restrict__ X, float* __restrict__ coords) {
  int node = blockIdx.x * blockDim.x + threadIdx.x;
  if (node >= NODES) return;
  const float* x = X + (size_t)node * 12;
  float nx = x[0], ny = x[1], nz = x[2];
  float cax = x[3], cay = x[4], caz = x[5];
  float cx = x[6], cy = x[7], cz = x[8];
  float ox = x[9], oy = x[10], oz = x[11];
  float bx = cax - nx, by = cay - ny, bz = caz - nz;    // bvec
  float vx = cx - cax, vy = cy - cay, vz = cz - caz;    // cvec
  float ax = by * vz - bz * vy;                         // avec = b x c
  float ay = bz * vx - bx * vz;
  float az = bx * vy - by * vx;
  float cbx = -0.58273431f * ax + 0.56802827f * bx - 0.54067466f * vx + cax;
  float cby = -0.58273431f * ay + 0.56802827f * by - 0.54067466f * vy + cay;
  float cbz = -0.58273431f * az + 0.56802827f * bz - 0.54067466f * vz + caz;
  float* o = coords + (size_t)node * 15;
  o[0] = nx;  o[1] = ny;  o[2] = nz;
  o[3] = cax; o[4] = cay; o[5] = caz;
  o[6] = cx;  o[7] = cy;  o[8] = cz;
  o[9] = ox;  o[10] = oy; o[11] = oz;
  o[12] = cbx; o[13] = cby; o[14] = cbz;
}

// ---------------------------------------------------------------------------
// 2. k-NN: per node, 32 smallest CA distances (tie -> lower index, matches
//    lax.top_k ordering). One 256-thread block per node; LDS argmin passes.
// ---------------------------------------------------------------------------
__global__ void knn_kernel(const float* __restrict__ coords,
                           const float* __restrict__ mask,
                           int* __restrict__ E_idx) {
  int node = blockIdx.x;                 // 0..NODES-1
  int b = node >> 10;                    // NNODE = 1024
  int t = threadIdx.x;                   // 0..255
  __shared__ float sd[NNODE];
  __shared__ float rv[256];
  __shared__ int   ri[256];
  float cax = coords[(size_t)node * 15 + 3];
  float cay = coords[(size_t)node * 15 + 4];
  float caz = coords[(size_t)node * 15 + 5];
  float mi = mask[node];
  for (int jj = t; jj < NNODE; jj += 256) {
    int jn = b * NNODE + jj;
    float dx = cax - coords[(size_t)jn * 15 + 3];
    float dy = cay - coords[(size_t)jn * 15 + 4];
    float dz = caz - coords[(size_t)jn * 15 + 5];
    float d = sqrtf(dx * dx + dy * dy + dz * dz + 1e-6f);
    sd[jj] = (mi * mask[jn] > 0.f) ? d : 1e6f;
  }
  __syncthreads();
  for (int sel = 0; sel < KNBR; ++sel) {
    float bv = 3e38f; int bidx = 0x7fffffff;
    for (int jj = t; jj < NNODE; jj += 256) {
      float v = sd[jj];
      if (v < bv || (v == bv && jj < bidx)) { bv = v; bidx = jj; }
    }
    rv[t] = bv; ri[t] = bidx;
    __syncthreads();
    for (int s = 128; s > 0; s >>= 1) {
      if (t < s) {
        float ov = rv[t + s]; int oi = ri[t + s];
        if (ov < rv[t] || (ov == rv[t] && oi < ri[t])) { rv[t] = ov; ri[t] = oi; }
      }
      __syncthreads();
    }
    if (t == 0) { E_idx[(size_t)node * KNBR + sel] = ri[0]; sd[ri[0]] = 3e38f; }
    __syncthreads();
  }
}

// ---------------------------------------------------------------------------
// 3. weight prep: W[K][N] f32 row-major -> Wt[N][K] f16 (column-major weights)
// ---------------------------------------------------------------------------
__global__ void wt_kernel(const float* __restrict__ W, _Float16* __restrict__ Wt,
                          int K, int N) {
  int idx = blockIdx.x * blockDim.x + threadIdx.x;
  if (idx >= K * N) return;
  int kk = idx / N, n = idx % N;
  Wt[(size_t)n * K + kk] = (_Float16)W[idx];
}

// ---------------------------------------------------------------------------
// 4. edge features: [pos(16) | rbf(400)] -> f16 (EC x 416). Thread = (edge,p),
//    p==0 -> positional embedding, p=1..25 -> one atom pair's 16 RBF bins.
// ---------------------------------------------------------------------------
__global__ void feat_kernel(const float* __restrict__ coords,
                            const int* __restrict__ E_idx,
                            const int* __restrict__ R_idx,
                            const int* __restrict__ chain,
                            const float* __restrict__ pos_W,
                            const float* __restrict__ pos_b,
                            _Float16* __restrict__ feat) {
  int idx = blockIdx.x * blockDim.x + threadIdx.x;
  if (idx >= EC * 26) return;
  int e = idx / 26;
  int p = idx % 26;
  int b = e >> 15;                 // / (NNODE*KNBR)
  int i = (e >> 5) & (NNODE - 1);
  int j = E_idx[e];
  _Float16* out = feat + (size_t)e * 416;
  if (p == 0) {
    int rI = R_idx[b * NNODE + i], rJ = R_idx[b * NNODE + j];
    int cI = chain[b * NNODE + i], cJ = chain[b * NNODE + j];
    int off = rI - rJ + 32;
    off = off < 0 ? 0 : (off > 64 ? 64 : off);
    int pidx = (cI == cJ) ? off : 65;
    #pragma unroll
    for (int t = 0; t < 16; ++t)
      out[t] = (_Float16)(pos_W[pidx * 16 + t] + pos_b[t]);
  } else {
    int pr = p - 1;
    int ai = pr / 5, aj = pr % 5;
    const float* ci = coords + ((size_t)(b * NNODE + i) * 5 + ai) * 3;
    const float* cj = coords + ((size_t)(b * NNODE + j) * 5 + aj) * 3;
    float dx = ci[0] - cj[0], dy = ci[1] - cj[1], dz = ci[2] - cj[2];
    float d = sqrtf(dx * dx + dy * dy + dz * dz + 1e-6f);
    #pragma unroll
    for (int t = 0; t < 16; ++t) {
      float mu = 2.0f + (20.0f / 15.0f) * (float)t;   // linspace(2,22,16)
      float z = (d - mu) * (1.0f / 1.25f);            // sigma = 20/16
      out[16 + pr * 16 + t] = (_Float16)expf(-z * z);
    }
  }
}

// ---------------------------------------------------------------------------
// 5. WMMA GEMM: out = act(A[MxKT]f16 @ W[KTxN] + bias), W as Wt[N][KT] f16.
//    Block = 8 waves; block stages a (NT*16)-col x KT weight tile into LDS
//    with async global->LDS DMA, then each wave computes a 16 x (NT*16)
//    output strip. KT compile-time -> fully unrolled, branch-free main loop:
//    per k-step, NT B-frags batch-loaded from LDS (padded stride, conflict-
//    free ds_load_b128), then NT WMMAs issued back-to-back.
//    M % 128 == 0, N % (NT*16) == 0. act: 0=none, 1=gelu.
// ---------------------------------------------------------------------------
template <int NT, int KT>
__global__ void gemm_wmma_kernel(const _Float16* __restrict__ A,
                                 const _Float16* __restrict__ Wt,
                                 const float* __restrict__ bias,
                                 int M, int N, int act,
                                 _Float16* __restrict__ outF16,
                                 float* __restrict__ outF32) {
  constexpr int CT = NT * 16;               // columns per block tile
  constexpr int LS = KT + 8;                // LDS row stride (8-half pad)
  extern __shared__ _Float16 ldsb[];        // CT x LS halves

  int lane = threadIdx.x & 31;
  int wave = threadIdx.x >> 5;
  int col0 = blockIdx.y * CT;

  // ---- stage weight tile: global -> LDS via ASYNCcnt-tracked DMA loads ----
  {
    constexpr int K8 = KT >> 3;             // v8h chunks per column
    for (int c = threadIdx.x; c < CT * K8; c += 256) {
      int n  = c / K8;
      int k8 = c - n * K8;
      unsigned ldsoff = (unsigned)(uintptr_t)&ldsb[n * LS + (k8 << 3)];
      const _Float16* g = Wt + (size_t)(col0 + n) * KT + (k8 << 3);
      asm volatile("global_load_async_to_lds_b128 %0, %1, off"
                   :: "v"(ldsoff), "v"(g) : "memory");
    }
    asm volatile("s_wait_asynccnt 0" ::: "memory");
  }
  __syncthreads();

  int strip = blockIdx.x * 8 + wave;        // 16-row strip
  int row0 = strip * 16;
  if (row0 >= M) return;                    // wave-uniform: EXEC stays full
  int m  = lane & 15;
  int hi = lane >> 4;

  v8f acc[NT] = {};
  const _Float16* Arow = A + (size_t)(row0 + m) * KT;
  const v8h* Ap = reinterpret_cast<const v8h*>(Arow);

  #pragma unroll
  for (int k = 0; k < KT; k += 32) {
    __builtin_prefetch(Arow + k + 256, 0, 0);         // global_prefetch_b8
    v8h a0 = Ap[(k >> 3) + hi];
    v8h a1 = Ap[(k >> 3) + 2 + hi];
    v16h afrag = __builtin_shufflevector(a0, a1, 0, 1, 2, 3, 4, 5, 6, 7,
                                                 8, 9, 10, 11, 12, 13, 14, 15);
    // batch all NT B-fragment loads, then burst the NT WMMAs
    v8h b0[NT], b1[NT];
    #pragma unroll
    for (int nt = 0; nt < NT; ++nt) {
      const v8h* Bp = reinterpret_cast<const v8h*>(
          &ldsb[(nt * 16 + m) * LS + k + hi * 8]);
      b0[nt] = Bp[0];
      b1[nt] = Bp[2];
    }
    #pragma unroll
    for (int nt = 0; nt < NT; ++nt) {
      v16h bfrag = __builtin_shufflevector(b0[nt], b1[nt],
                                           0, 1, 2, 3, 4, 5, 6, 7,
                                           8, 9, 10, 11, 12, 13, 14, 15);
      acc[nt] = __builtin_amdgcn_wmma_f32_16x16x32_f16(
          false, afrag, false, bfrag, (short)0, acc[nt], false, false);
    }
  }

  #pragma unroll
  for (int nt = 0; nt < NT; ++nt) {
    int c = col0 + nt * 16 + m;
    float bv = bias ? bias[c] : 0.0f;
    #pragma unroll
    for (int r = 0; r < 8; ++r) {
      int row = row0 + r + hi * 8;          // C layout: VGPR r -> M = r + 8*hi
      float v = acc[nt][r] + bv;
      if (act == 1) v = gelu_f(v);
      size_t o = (size_t)row * N + c;
      if (outF32) outF32[o] = v;
      if (outF16) outF16[o] = (_Float16)v;
    }
  }
}

// ---------------------------------------------------------------------------
// 6. LayerNorm over 128 channels: y = LN(x [+res]) * g + b [* mask[row]]
// ---------------------------------------------------------------------------
__global__ void ln_kernel(const float* __restrict__ x,
                          const float* __restrict__ res,
                          const float* __restrict__ g,
                          const float* __restrict__ bta,
                          const float* __restrict__ rowmask,
                          float* __restrict__ outF32,
                          _Float16* __restrict__ outF16) {
  int row = blockIdx.x;
  int t = threadIdx.x;                      // 128
  __shared__ float red[128];
  __shared__ float s_mean, s_rstd;
  size_t o = (size_t)row * 128 + t;
  float v = x[o] + (res ? res[o] : 0.0f);
  red[t] = v;
  __syncthreads();
  for (int s = 64; s > 0; s >>= 1) { if (t < s) red[t] += red[t + s]; __syncthreads(); }
  if (t == 0) s_mean = red[0] * (1.0f / 128.0f);
  __syncthreads();
  float dev = v - s_mean;
  red[t] = dev * dev;
  __syncthreads();
  for (int s = 64; s > 0; s >>= 1) { if (t < s) red[t] += red[t + s]; __syncthreads(); }
  if (t == 0) s_rstd = rsqrtf(red[0] * (1.0f / 128.0f) + 1e-5f);
  __syncthreads();
  float y = dev * s_rstd * g[t] + bta[t];
  if (rowmask) y *= rowmask[row];
  if (outF32) outF32[o] = y;
  if (outF16) outF16[o] = (_Float16)y;
}

// ---------------------------------------------------------------------------
// 7. hEV = concat([hV_i, hE, hV_j]) -> f16 (EC x 384)
// ---------------------------------------------------------------------------
__global__ void build_hEV_kernel(const float* __restrict__ hV,
                                 const float* __restrict__ hE,
                                 const int* __restrict__ E_idx,
                                 _Float16* __restrict__ hEV) {
  int idx = blockIdx.x * blockDim.x + threadIdx.x;
  if (idx >= EC * 384) return;
  int e = idx / 384;
  int c = idx % 384;
  int b = e >> 15;
  int i = (e >> 5) & (NNODE - 1);
  float v;
  if (c < 128)      v = hV[(size_t)(b * NNODE + i) * 128 + c];
  else if (c < 256) v = hE[(size_t)e * 128 + (c - 128)];
  else              v = hV[(size_t)(b * NNODE + E_idx[e]) * 128 + (c - 256)];
  hEV[idx] = (_Float16)v;
}

// ---------------------------------------------------------------------------
// 8. node update 1: hV = LN(hV + sum_k(msg * m_att)/30) * g + b
// ---------------------------------------------------------------------------
__global__ void node_aggregate_ln_kernel(const float* __restrict__ msg,
                                         const float* __restrict__ hV_in,
                                         const float* __restrict__ mask,
                                         const int* __restrict__ E_idx,
                                         const float* __restrict__ g,
                                         const float* __restrict__ bta,
                                         float* __restrict__ hV_out,
                                         _Float16* __restrict__ hV16) {
  int node = blockIdx.x;
  int t = threadIdx.x;                      // 128
  int b = node >> 10;
  __shared__ float matt[KNBR];
  __shared__ float red[128];
  __shared__ float s_mean, s_rstd;
  if (t < KNBR) {
    int j = E_idx[(size_t)node * KNBR + t];
    matt[t] = mask[node] * mask[b * NNODE + j];
  }
  __syncthreads();
  float s = 0.0f;
  for (int kk = 0; kk < KNBR; ++kk)
    s += msg[((size_t)node * KNBR + kk) * 128 + t] * matt[kk];
  float v = hV_in[(size_t)node * 128 + t] + s * (1.0f / 30.0f);
  red[t] = v;
  __syncthreads();
  for (int st = 64; st > 0; st >>= 1) { if (t < st) red[t] += red[t + st]; __syncthreads(); }
  if (t == 0) s_mean = red[0] * (1.0f / 128.0f);
  __syncthreads();
  float dev = v - s_mean;
  red[t] = dev * dev;
  __syncthreads();
  for (int st = 64; st > 0; st >>= 1) { if (t < st) red[t] += red[t + st]; __syncthreads(); }
  if (t == 0) s_rstd = rsqrtf(red[0] * (1.0f / 128.0f) + 1e-5f);
  __syncthreads();
  float y = dev * s_rstd * g[t] + bta[t];
  hV_out[(size_t)node * 128 + t] = y;
  hV16[(size_t)node * 128 + t] = (_Float16)y;
}

// ---------------------------------------------------------------------------
// host: orchestrate the whole pipeline on `stream`
// ---------------------------------------------------------------------------
extern "C" void kernel_launch(void* const* d_in, const int* in_sizes, int n_in,
                              void* d_out, int out_size, void* d_ws, size_t ws_size,
                              hipStream_t stream) {
  (void)in_sizes; (void)n_in; (void)out_size; (void)ws_size;
  const float* X      = (const float*)d_in[0];
  const float* mask   = (const float*)d_in[1];
  const float* pos_W  = (const float*)d_in[2];
  const float* pos_b  = (const float*)d_in[3];
  const float* We_W   = (const float*)d_in[4];
  const float* We_b   = (const float*)d_in[5];
  const float* ne_g   = (const float*)d_in[6];
  const float* ne_b   = (const float*)d_in[7];
  const float* W1     = (const float*)d_in[8];
  const float* b1     = (const float*)d_in[9];
  const float* W2     = (const float*)d_in[10];
  const float* b2     = (const float*)d_in[11];
  const float* W3     = (const float*)d_in[12];
  const float* b3     = (const float*)d_in[13];
  const float* n1_g   = (const float*)d_in[14];
  const float* n1_b   = (const float*)d_in[15];
  const float* Wi     = (const float*)d_in[16];
  const float* bi     = (const float*)d_in[17];
  const float* Wo     = (const float*)d_in[18];
  const float* bo     = (const float*)d_in[19];
  const float* n2_g   = (const float*)d_in[20];
  const float* n2_b   = (const float*)d_in[21];
  const float* W11    = (const float*)d_in[22];
  const float* b11    = (const float*)d_in[23];
  const float* W12    = (const float*)d_in[24];
  const float* b12    = (const float*)d_in[25];
  const float* W13    = (const float*)d_in[26];
  const float* b13    = (const float*)d_in[27];
  const float* n3_g   = (const float*)d_in[28];
  const float* n3_b   = (const float*)d_in[29];
  const float* fln_g  = (const float*)d_in[30];
  const float* fln_b  = (const float*)d_in[31];
  const float* fd_W   = (const float*)d_in[32];
  const float* fd_b   = (const float*)d_in[33];
  const int*   R_idx  = (const int*)d_in[34];
  const int*   chain  = (const int*)d_in[35];

  // -------- workspace arena --------
  char* ws = (char*)d_ws;
  size_t off = 0;
  auto arena = [&](size_t bytes) -> char* {
    char* p = ws + off;
    off = (off + bytes + 255) & ~(size_t)255;
    return p;
  };
  float*     coords = (float*)arena((size_t)NODES * 15 * 4);
  int*       E_idx  = (int*)arena((size_t)EC * 4);
  _Float16*  warena = (_Float16*)arena((size_t)905216 * 2);
  _Float16*  feat   = (_Float16*)arena((size_t)EC * 416 * 2); // hEV aliases this
  float*     hE     = (float*)arena((size_t)EC * 128 * 4);
  _Float16*  t1     = (_Float16*)arena((size_t)EC * 128 * 2);
  _Float16*  t2     = (_Float16*)arena((size_t)EC * 128 * 2);
  float*     msg    = (float*)arena((size_t)EC * 128 * 4);
  float*     hV     = (float*)arena((size_t)NODES * 128 * 4);
  _Float16*  hV16   = (_Float16*)arena((size_t)NODES * 128 * 2);
  _Float16*  ffn1   = (_Float16*)arena((size_t)NODES * 512 * 2);
  float*     ffn2   = (float*)arena((size_t)NODES * 128 * 4);
  _Float16*  hvln16 = (_Float16*)arena((size_t)NODES * 128 * 2);
  _Float16*  hEV    = feat;  // feat consumed before first hEV build

  // f16 transposed weight sub-arena
  _Float16* wtWe  = warena;
  _Float16* wtW1  = wtWe  + 53248;   // 416*128
  _Float16* wtW2  = wtW1  + 147456;  // 3*384*128
  _Float16* wtW3  = wtW2  + 49152;   // 3*128*128
  _Float16* wtWi  = wtW3  + 49152;
  _Float16* wtWo  = wtWi  + 196608;  // 3*512*128
  _Float16* wtW11 = wtWo  + 196608;
  _Float16* wtW12 = wtW11 + 98304;   // 2*384*128
  _Float16* wtW13 = wtW12 + 32768;
  _Float16* wtFd  = wtW13 + 32768;   // 384*128

  auto wt = [&](const float* src, _Float16* dst, int K, int N) {
    int tot = K * N;
    wt_kernel<<<dim3((tot + 255) / 256), dim3(256), 0, stream>>>(src, dst, K, N);
  };
  // Compile-time K dispatch. NT=4 (64-col tile) keeps dynamic LDS <= 54.3 KB
  // for K<=416; the K=512 GEMM drops to NT=2 (32-col tile, 33.3 KB).
  auto gemm = [&](const _Float16* A, const _Float16* Wt, const float* bias,
                  int M, int N, int K, int act, _Float16* o16, float* o32) {
    if (K == 384) {
      gemm_wmma_kernel<4, 384><<<dim3(M / 128, N / 64), dim3(256),
          64 * (384 + 8) * sizeof(_Float16), stream>>>(A, Wt, bias, M, N, act, o16, o32);
    } else if (K == 416) {
      gemm_wmma_kernel<4, 416><<<dim3(M / 128, N / 64), dim3(256),
          64 * (416 + 8) * sizeof(_Float16), stream>>>(A, Wt, bias, M, N, act, o16, o32);
    } else if (K == 128) {
      gemm_wmma_kernel<4, 128><<<dim3(M / 128, N / 64), dim3(256),
          64 * (128 + 8) * sizeof(_Float16), stream>>>(A, Wt, bias, M, N, act, o16, o32);
    } else { // K == 512
      gemm_wmma_kernel<2, 512><<<dim3(M / 128, N / 32), dim3(256),
          32 * (512 + 8) * sizeof(_Float16), stream>>>(A, Wt, bias, M, N, act, o16, o32);
    }
  };
  auto ln = [&](const float* x, const float* res, const float* g, const float* b,
                const float* rmask, int rows, float* o32, _Float16* o16) {
    ln_kernel<<<dim3(rows), dim3(128), 0, stream>>>(x, res, g, b, rmask, o32, o16);
  };

  // -------- geometry + graph --------
  coords_kernel<<<dim3((NODES + 255) / 256), dim3(256), 0, stream>>>(X, coords);
  knn_kernel<<<dim3(NODES), dim3(256), 0, stream>>>(coords, mask, E_idx);

  // -------- weight prep (f32 -> f16, transposed) --------
  wt(We_W, wtWe, 416, 128);
  for (int l = 0; l < 3; ++l) {
    wt(W1 + (size_t)l * 384 * 128, wtW1 + (size_t)l * 49152, 384, 128);
    wt(W2 + (size_t)l * 128 * 128, wtW2 + (size_t)l * 16384, 128, 128);
    wt(W3 + (size_t)l * 128 * 128, wtW3 + (size_t)l * 16384, 128, 128);
    wt(Wi + (size_t)l * 128 * 512, wtWi + (size_t)l * 65536, 128, 512);
    wt(Wo + (size_t)l * 512 * 128, wtWo + (size_t)l * 65536, 512, 128);
  }
  for (int l = 0; l < 2; ++l) {
    wt(W11 + (size_t)l * 384 * 128, wtW11 + (size_t)l * 49152, 384, 128);
    wt(W12 + (size_t)l * 128 * 128, wtW12 + (size_t)l * 16384, 128, 128);
    wt(W13 + (size_t)l * 128 * 128, wtW13 + (size_t)l * 16384, 128, 128);
  }
  wt(fd_W, wtFd, 128, 384);

  // -------- edge encoder: hE = LN(feat @ We + We_b) --------
  feat_kernel<<<dim3((EC * 26 + 255) / 256), dim3(256), 0, stream>>>(
      coords, E_idx, R_idx, chain, pos_W, pos_b, feat);
  gemm(feat, wtWe, We_b, EC, 128, 416, 0, nullptr, msg);
  ln(msg, nullptr, ne_g, ne_b, nullptr, EC, hE, nullptr);

  hipMemsetAsync(hV, 0, (size_t)NODES * 128 * 4, stream);

  // -------- 3 message-passing layers --------
  for (int l = 0; l < 3; ++l) {
    build_hEV_kernel<<<dim3((EC * 384 + 255) / 256), dim3(256), 0, stream>>>(
        hV, hE, E_idx, hEV);
    gemm(hEV, wtW1 + (size_t)l * 49152, b1 + l * 128, EC, 128, 384, 1, t1, nullptr);
    gemm(t1,  wtW2 + (size_t)l * 16384, b2 + l * 128, EC, 128, 128, 1, t2, nullptr);
    gemm(t2,  wtW3 + (size_t)l * 16384, b3 + l * 128, EC, 128, 128, 0, nullptr, msg);
    node_aggregate_ln_kernel<<<dim3(NODES), dim3(128), 0, stream>>>(
        msg, hV, mask, E_idx, n1_g + l * 128, n1_b + l * 128, hV, hV16);
    // FFN: hV = LN(hV + gelu(hV@Wi+bi)@Wo+bo) * mask
    gemm(hV16, wtWi + (size_t)l * 65536, bi + l * 512, NODES, 512, 128, 1, ffn1, nullptr);
    gemm(ffn1, wtWo + (size_t)l * 65536, bo + l * 128, NODES, 128, 512, 0, nullptr, ffn2);
    ln(ffn2, hV, n2_g + l * 128, n2_b + l * 128, mask, NODES, hV, hV16);
    if (l < 2) {
      build_hEV_kernel<<<dim3((EC * 384 + 255) / 256), dim3(256), 0, stream>>>(
          hV, hE, E_idx, hEV);
      gemm(hEV, wtW11 + (size_t)l * 49152, b11 + l * 128, EC, 128, 384, 1, t1, nullptr);
      gemm(t1,  wtW12 + (size_t)l * 16384, b12 + l * 128, EC, 128, 128, 1, t2, nullptr);
      gemm(t2,  wtW13 + (size_t)l * 16384, b13 + l * 128, EC, 128, 128, 0, nullptr, msg);
      ln(msg, hE, n3_g + l * 128, n3_b + l * 128, nullptr, EC, hE, nullptr);
    }
  }

  // -------- head: out = LN(hV) @ fd_W + fd_b --------
  ln(hV, nullptr, fln_g, fln_b, nullptr, NODES, nullptr, hvln16);
  gemm(hvln16, wtFd, fd_b, NODES, 384, 128, 0, nullptr, (float*)d_out);
}
